// RNN_37426345017473
// MI455X (gfx1250) — matrix operand order repaired and checked by reading, not compiled, not run on hardware
//
#include <hip/hip_runtime.h>
#include <hip/hip_bf16.h>

// RNN: h_t = tanh(x_t @ Wx^T + h_{t-1} @ Wh^T + b), T=B=256, D=1024.
// Phase 1: Xp = x @ Wx^T + b  -> stored into d_out[(b,t,:)] (f32).
// Phase 2: per t: out[b,t,:] = tanh(out[b,t,:] + h_{t-1} @ Wh^T).
// Fast path (ws >= 5MB): W pre-converted to f16 in ws; h_t kept as f16
// ping-pong in ws; tiles streamed with GLOBAL_LOAD_ASYNC_TO_LDS_B128 into a
// 3-stage LDS pipeline with partial s_wait_asynccnt (2 stages stay in flight
// behind the WMMAs). Fallback path: ws-free f32->f16 conversion through LDS.

#define T_STEPS 256
#define BATCH   256
#define DIM     1024
#define LDW     2048          // W is (D, 2D) row-major
#define NK      (DIM / BK)    // 32 K-slices

#define BM 128
#define BN 128
#define BK 32
#define LDS_STRIDE 40         // 32 halves + 8 pad -> conflict-free b128 LDS reads
#define NSTAGE 3

typedef __attribute__((ext_vector_type(16))) _Float16 v16h;
typedef __attribute__((ext_vector_type(8)))  _Float16 v8h;
typedef __attribute__((ext_vector_type(4)))  _Float16 v4h;
typedef __attribute__((ext_vector_type(8)))  float    v8f;
typedef __attribute__((ext_vector_type(4)))  float    v4f;

// Wait until at most `imm` async instructions (per wave) remain outstanding.
// ASYNCcnt retires in order, so <=N means "everything but the newest N is done".
#define ASYNC_WAIT_LE(imm) asm volatile("s_wait_asynccnt " #imm ::: "memory")

// ---- LDS tile fill: 128 rows x 32 cols, f32 global -> f16 LDS --------------
__device__ __forceinline__ void load_tile_f32(_Float16* __restrict__ dst,
                                              const float* __restrict__ src,
                                              int ld, int tid) {
#pragma unroll
  for (int i = 0; i < 4; ++i) {
    int idx = tid + i * 256;            // 0..1023 quads
    int row = idx >> 3;                 // 0..127
    int qc  = (idx & 7) << 2;           // 0,4,...,28
    v4f v = *(const v4f*)(src + (long)row * ld + qc);
    v4h h = { (_Float16)v.x, (_Float16)v.y, (_Float16)v.z, (_Float16)v.w };
    *(v4h*)(dst + row * LDS_STRIDE + qc) = h;
  }
}

__device__ __forceinline__ void zero_tile(_Float16* __restrict__ dst, int tid) {
#pragma unroll
  for (int i = 0; i < 4; ++i) {
    int idx = tid + i * 256;
    int row = idx >> 3;
    int qc  = (idx & 7) << 2;
    *(v4h*)(dst + row * LDS_STRIDE + qc) = (v4h){ (_Float16)0, (_Float16)0, (_Float16)0, (_Float16)0 };
  }
}

// ---- async f16 tile fill: 128 rows x 32 halves, 512 x 16B chunks -----------
// GLOBAL_LOAD_ASYNC_TO_LDS_B128: per-lane LDS byte addr in VDST (low 32 bits
// of generic shared pointer == LDS offset), 64-bit global addr in VGPR pair.
// 2 async instructions per wave per tile.
__device__ __forceinline__ void async_tile_f16(const _Float16* lds_tile,
                                               const _Float16* __restrict__ src,
                                               int ld, int tid) {
  const unsigned base = (unsigned)(uintptr_t)lds_tile;
#pragma unroll
  for (int i = 0; i < 2; ++i) {
    int c   = tid + (i << 8);           // 0..511 chunks
    int row = c >> 2;                   // 0..127
    int cc  = (c & 3) << 3;             // half offset: 0,8,16,24
    unsigned lds = base + (unsigned)((row * LDS_STRIDE + cc) << 1);
    unsigned long long ga =
        (unsigned long long)(uintptr_t)(src + (long)row * ld + cc);
    asm volatile("global_load_async_to_lds_b128 %0, %1, off"
                 :: "v"(lds), "v"(ga) : "memory");
  }
}

// ---- A fragment (16x32 f16): lanes 0-15 hold K={0..7,16..23}, 16-31 K+8 ----
__device__ __forceinline__ v16h load_afrag(const _Float16* __restrict__ tile,
                                           int row, int lane) {
  const int kb = (lane & 16) ? 8 : 0;
  const _Float16* p = tile + row * LDS_STRIDE + kb;
  v16h f;
  ((v8h*)&f)[0] = *(const v8h*)(p);
  ((v8h*)&f)[1] = *(const v8h*)(p + 16);
  return f;
}

// ---- B fragment (32x16 f16): lanes 0-15 hold K={0..7,16..23}, 16-31 K+8 ----
__device__ __forceinline__ v16h load_bfrag(const _Float16* __restrict__ tile,
                                           int row, int lane) {
  const int kb = (lane & 16) ? 8 : 0;
  const _Float16* p = tile + row * LDS_STRIDE + kb;
  v16h f;
  ((v8h*)&f)[0] = *(const v8h*)(p);
  ((v8h*)&f)[1] = *(const v8h*)(p + 16);
  return f;
}

// ============================================================================
// W f32 (D x 2D) -> f16
// ============================================================================
__global__ __launch_bounds__(256)
void rnn_convert_w(const float* __restrict__ W, _Float16* __restrict__ W16) {
  const long i = ((long)blockIdx.x * 256 + threadIdx.x) * 4;
  v4f v = *(const v4f*)(W + i);
  v4h h = { (_Float16)v.x, (_Float16)v.y, (_Float16)v.z, (_Float16)v.w };
  *(v4h*)(W16 + i) = h;
}

// Zero the h_{-1} f16 buffer (h0 = 0 in the reference scan).
__global__ __launch_bounds__(256)
void rnn_zero_h(_Float16* __restrict__ H) {
  const long i = ((long)blockIdx.x * 256 + threadIdx.x) * 8;
  *(v8h*)(H + i) = (v8h){ (_Float16)0, (_Float16)0, (_Float16)0, (_Float16)0,
                          (_Float16)0, (_Float16)0, (_Float16)0, (_Float16)0 };
}

// ============================================================================
// FAST Phase 1: out[(b,t,:)] = x[(t,b,:)] @ Wx16^T + bias
// A: f32 global -> cvt -> LDS; B: f16 async -> LDS; 3-stage pipeline,
// s_wait_asynccnt <= 2 keeps the newest B stage in flight across the barrier.
// ============================================================================
__global__ __launch_bounds__(256)
void rnn_phase1_fast(const float* __restrict__ x, const _Float16* __restrict__ W16,
                     const float* __restrict__ bias, float* __restrict__ out) {
  __shared__ _Float16 sA[NSTAGE][BM * LDS_STRIDE];
  __shared__ _Float16 sB[NSTAGE][BN * LDS_STRIDE];

  const int tid  = threadIdx.x;
  const int lane = tid & 31;
  const int wave = tid >> 5;
  const int wm   = wave >> 2;
  const int wn   = wave & 3;
  const long m0  = (long)blockIdx.x * BM;
  const int  n0  = blockIdx.y * BN;

  const float*    Ablk = x + m0 * DIM;
  const _Float16* Bblk = W16 + (long)n0 * LDW;     // Wx columns k in [0,D)

  v8f acc[4][2] = {};

  // Prologue: stages 0 and 1 in flight.
  load_tile_f32(sA[0], Ablk, DIM, tid);
  async_tile_f16(sB[0], Bblk, LDW, tid);
  load_tile_f32(sA[1], Ablk + BK, DIM, tid);
  async_tile_f16(sB[1], Bblk + BK, LDW, tid);

  int buf = 0;
  for (int kt = 0; kt < NK; ++kt) {
    ASYNC_WAIT_LE(0x2);        // stage kt complete; stage kt+1 may be in flight
    __syncthreads();           // stage kt visible to all waves
    if (kt + 2 < NK) {         // issue stage kt+2 (buffer read at kt-1 is free)
      int nb = buf + 2; if (nb >= NSTAGE) nb -= NSTAGE;
      load_tile_f32(sA[nb], Ablk + (kt + 2) * BK, DIM, tid);
      async_tile_f16(sB[nb], Bblk + (kt + 2) * BK, LDW, tid);
    }

    v16h bfr[2];
#pragma unroll
    for (int nt = 0; nt < 2; ++nt)
      bfr[nt] = load_bfrag(sB[buf], wn * 32 + nt * 16 + (lane & 15), lane);
#pragma unroll
    for (int mt = 0; mt < 4; ++mt) {
      v16h afr = load_afrag(sA[buf], wm * 64 + mt * 16 + (lane & 15), lane);
#pragma unroll
      for (int nt = 0; nt < 2; ++nt)
        acc[mt][nt] = __builtin_amdgcn_wmma_f32_16x16x32_f16(
            false, afr, false, bfr[nt], (short)0, acc[mt][nt], false, false);
    }
    if (++buf == NSTAGE) buf = 0;
  }

  const int mhalf = (lane & 16) ? 8 : 0;
  const int ncol  = lane & 15;
#pragma unroll
  for (int nt = 0; nt < 2; ++nt) {
    const int n = n0 + wn * 32 + nt * 16 + ncol;
    const float bv = bias[n];
#pragma unroll
    for (int mt = 0; mt < 4; ++mt) {
#pragma unroll
      for (int r = 0; r < 8; ++r) {
        const long m = m0 + wm * 64 + mt * 16 + r + mhalf;
        const long t = m >> 8;
        const long b = m & (BATCH - 1);
        out[(b * T_STEPS + t) * DIM + n] = acc[mt][nt][r] + bv;
      }
    }
  }
}

// ============================================================================
// FAST Phase 2 (one launch per t):
//   out[b,t,:] = tanh(out[b,t,:] + h_{t-1} @ Wh16^T);  Hcur16[b,:] = (f16)h_t
// A: f16 async from Hprev16 (pre-zeroed buffer at t==0); B: f16 async from
// Wh16. 4 async instrs per stage -> s_wait_asynccnt <= 4.
// ============================================================================
__global__ __launch_bounds__(256)
void rnn_phase2_fast(const _Float16* __restrict__ W16,
                     const _Float16* __restrict__ Hprev16,
                     _Float16* __restrict__ Hcur16,
                     float* __restrict__ out, int t) {
  __shared__ _Float16 sA[NSTAGE][BM * LDS_STRIDE];
  __shared__ _Float16 sB[NSTAGE][BN * LDS_STRIDE];

  const int tid  = threadIdx.x;
  const int lane = tid & 31;
  const int wave = tid >> 5;
  const int wm   = wave >> 2;
  const int wn   = wave & 3;
  const long m0  = (long)blockIdx.x * BM;          // batch rows
  const int  n0  = blockIdx.y * BN;

  const _Float16* Ablk = Hprev16 + m0 * DIM;
  const _Float16* Bblk = W16 + (long)n0 * LDW + DIM;   // Wh = W[:, D:2D]

  const int mhalf = (lane & 16) ? 8 : 0;
  const int ncol  = lane & 15;

  // Init accumulators from pre-stored Xp (already includes bias).
  v8f acc[4][2];
#pragma unroll
  for (int mt = 0; mt < 4; ++mt)
#pragma unroll
    for (int nt = 0; nt < 2; ++nt) {
      const int n = n0 + wn * 32 + nt * 16 + ncol;
#pragma unroll
      for (int r = 0; r < 8; ++r) {
        const long b = m0 + wm * 64 + mt * 16 + r + mhalf;
        acc[mt][nt][r] = out[(b * T_STEPS + t) * DIM + n];
      }
    }

  // Prologue: stages 0 and 1 in flight.
  async_tile_f16(sA[0], Ablk, DIM, tid);
  async_tile_f16(sB[0], Bblk, LDW, tid);
  async_tile_f16(sA[1], Ablk + BK, DIM, tid);
  async_tile_f16(sB[1], Bblk + BK, LDW, tid);

  int buf = 0;
  for (int kt = 0; kt < NK; ++kt) {
    ASYNC_WAIT_LE(0x4);        // stage kt complete; stage kt+1 may be in flight
    __syncthreads();
    if (kt + 2 < NK) {
      int nb = buf + 2; if (nb >= NSTAGE) nb -= NSTAGE;
      async_tile_f16(sA[nb], Ablk + (kt + 2) * BK, DIM, tid);
      async_tile_f16(sB[nb], Bblk + (kt + 2) * BK, LDW, tid);
    }

    v16h bfr[2];
#pragma unroll
    for (int nt = 0; nt < 2; ++nt)
      bfr[nt] = load_bfrag(sB[buf], wn * 32 + nt * 16 + (lane & 15), lane);
#pragma unroll
    for (int mt = 0; mt < 4; ++mt) {
      v16h afr = load_afrag(sA[buf], wm * 64 + mt * 16 + (lane & 15), lane);
#pragma unroll
      for (int nt = 0; nt < 2; ++nt)
        acc[mt][nt] = __builtin_amdgcn_wmma_f32_16x16x32_f16(
            false, afr, false, bfr[nt], (short)0, acc[mt][nt], false, false);
    }
    if (++buf == NSTAGE) buf = 0;
  }

#pragma unroll
  for (int nt = 0; nt < 2; ++nt) {
    const int n = n0 + wn * 32 + nt * 16 + ncol;
#pragma unroll
    for (int mt = 0; mt < 4; ++mt) {
#pragma unroll
      for (int r = 0; r < 8; ++r) {
        const long b = m0 + wm * 64 + mt * 16 + r + mhalf;
        const float h = tanhf(acc[mt][nt][r]);
        out[(b * T_STEPS + t) * DIM + n] = h;
        Hcur16[b * DIM + n] = (_Float16)h;
      }
    }
  }
}

// ============================================================================
// FALLBACK (ws-free) kernels
// ============================================================================
__global__ __launch_bounds__(256)
void rnn_phase1_xw(const float* __restrict__ x, const float* __restrict__ W,
                   const float* __restrict__ bias, float* __restrict__ out) {
  __shared__ _Float16 sA[BM * LDS_STRIDE];
  __shared__ _Float16 sB[BN * LDS_STRIDE];

  const int tid  = threadIdx.x;
  const int lane = tid & 31;
  const int wave = tid >> 5;
  const int wm   = wave >> 2;
  const int wn   = wave & 3;
  const long m0  = (long)blockIdx.x * BM;
  const int  n0  = blockIdx.y * BN;

  const float* Ablk = x + m0 * DIM;
  const float* Bblk = W + (long)n0 * LDW;

  v8f acc[4][2] = {};

  for (int k0 = 0; k0 < DIM; k0 += BK) {
    load_tile_f32(sA, Ablk + k0, DIM, tid);
    load_tile_f32(sB, Bblk + k0, LDW, tid);
    __syncthreads();
    v16h bfr[2];
#pragma unroll
    for (int nt = 0; nt < 2; ++nt)
      bfr[nt] = load_bfrag(sB, wn * 32 + nt * 16 + (lane & 15), lane);
#pragma unroll
    for (int mt = 0; mt < 4; ++mt) {
      v16h afr = load_afrag(sA, wm * 64 + mt * 16 + (lane & 15), lane);
#pragma unroll
      for (int nt = 0; nt < 2; ++nt)
        acc[mt][nt] = __builtin_amdgcn_wmma_f32_16x16x32_f16(
            false, afr, false, bfr[nt], (short)0, acc[mt][nt], false, false);
    }
    __syncthreads();
  }

  const int mhalf = (lane & 16) ? 8 : 0;
  const int ncol  = lane & 15;
#pragma unroll
  for (int nt = 0; nt < 2; ++nt) {
    const int n = n0 + wn * 32 + nt * 16 + ncol;
    const float bv = bias[n];
#pragma unroll
    for (int mt = 0; mt < 4; ++mt) {
#pragma unroll
      for (int r = 0; r < 8; ++r) {
        const long m = m0 + wm * 64 + mt * 16 + r + mhalf;
        const long t = m >> 8;
        const long b = m & (BATCH - 1);
        out[(b * T_STEPS + t) * DIM + n] = acc[mt][nt][r] + bv;
      }
    }
  }
}

__global__ __launch_bounds__(256)
void rnn_phase2_hw(const float* __restrict__ W, float* __restrict__ out, int t) {
  __shared__ _Float16 sA[BM * LDS_STRIDE];
  __shared__ _Float16 sB[BN * LDS_STRIDE];

  const int tid  = threadIdx.x;
  const int lane = tid & 31;
  const int wave = tid >> 5;
  const int wm   = wave >> 2;
  const int wn   = wave & 3;
  const long m0  = (long)blockIdx.x * BM;
  const int  n0  = blockIdx.y * BN;

  const int   ldA  = T_STEPS * DIM;
  const float* Ablk = out + m0 * (long)ldA + (long)(t - 1) * DIM;
  const float* Bblk = W + (long)n0 * LDW + DIM;

  const int mhalf = (lane & 16) ? 8 : 0;
  const int ncol  = lane & 15;

  v8f acc[4][2];
#pragma unroll
  for (int mt = 0; mt < 4; ++mt)
#pragma unroll
    for (int nt = 0; nt < 2; ++nt) {
      const int n = n0 + wn * 32 + nt * 16 + ncol;
#pragma unroll
      for (int r = 0; r < 8; ++r) {
        const long b = m0 + wm * 64 + mt * 16 + r + mhalf;
        acc[mt][nt][r] = out[(b * T_STEPS + t) * DIM + n];
      }
    }

  for (int k0 = 0; k0 < DIM; k0 += BK) {
    if (t == 0) zero_tile(sA, tid);
    else        load_tile_f32(sA, Ablk + k0, ldA, tid);
    load_tile_f32(sB, Bblk + k0, LDW, tid);
    __syncthreads();
    v16h bfr[2];
#pragma unroll
    for (int nt = 0; nt < 2; ++nt)
      bfr[nt] = load_bfrag(sB, wn * 32 + nt * 16 + (lane & 15), lane);
#pragma unroll
    for (int mt = 0; mt < 4; ++mt) {
      v16h afr = load_afrag(sA, wm * 64 + mt * 16 + (lane & 15), lane);
#pragma unroll
      for (int nt = 0; nt < 2; ++nt)
        acc[mt][nt] = __builtin_amdgcn_wmma_f32_16x16x32_f16(
            false, afr, false, bfr[nt], (short)0, acc[mt][nt], false, false);
    }
    __syncthreads();
  }

#pragma unroll
  for (int nt = 0; nt < 2; ++nt) {
    const int n = n0 + wn * 32 + nt * 16 + ncol;
#pragma unroll
    for (int mt = 0; mt < 4; ++mt) {
#pragma unroll
      for (int r = 0; r < 8; ++r) {
        const long b = m0 + wm * 64 + mt * 16 + r + mhalf;
        out[(b * T_STEPS + t) * DIM + n] = tanhf(acc[mt][nt][r]);
      }
    }
  }
}

// ============================================================================
extern "C" void kernel_launch(void* const* d_in, const int* in_sizes, int n_in,
                              void* d_out, int out_size, void* d_ws, size_t ws_size,
                              hipStream_t stream) {
  (void)in_sizes; (void)n_in; (void)out_size;
  const float* x    = (const float*)d_in[0];   // (T, B, D) f32
  const float* W    = (const float*)d_in[1];   // (D, 2D)  f32
  const float* bias = (const float*)d_in[2];   // (D,)     f32
  float*       out  = (float*)d_out;           // (B, T, D) f32

  const size_t W16_BYTES = (size_t)DIM * LDW * sizeof(_Float16);        // 4 MB
  const size_t H_BYTES   = (size_t)BATCH * DIM * sizeof(_Float16);      // 512 KB
  const size_t WS_NEED   = W16_BYTES + 2 * H_BYTES;                     // 5 MB

  dim3 blk(256);
  if (d_ws != nullptr && ws_size >= WS_NEED) {
    _Float16* W16 = (_Float16*)d_ws;
    _Float16* H16 = (_Float16*)((char*)d_ws + W16_BYTES);
    const long h_elems = (long)BATCH * DIM;

    rnn_convert_w<<<dim3(2048), blk, 0, stream>>>(W, W16);
    rnn_zero_h<<<dim3(128), blk, 0, stream>>>(H16 + 1 * h_elems);  // h_{-1} = 0
    rnn_phase1_fast<<<dim3(512, 8), blk, 0, stream>>>(x, W16, bias, out);
    for (int t = 0; t < T_STEPS; ++t) {
      const _Float16* Hprev = H16 + ((t + 1) & 1) * h_elems;
      _Float16*       Hcur  = H16 + (t & 1) * h_elems;
      rnn_phase2_fast<<<dim3(2, 8), blk, 0, stream>>>(W16, Hprev, Hcur, out, t);
    }
  } else {
    rnn_phase1_xw<<<dim3(512, 8), blk, 0, stream>>>(x, W, bias, out);
    for (int t = 0; t < T_STEPS; ++t)
      rnn_phase2_hw<<<dim3(2, 8), blk, 0, stream>>>(W, out, t);
  }
}
